// ScenePredNet_3006477107425
// MI455X (gfx1250) — matrix operand order repaired and checked by reading, not compiled, not run on hardware
//
#include <hip/hip_runtime.h>
#include <hip/hip_bf16.h>

typedef __attribute__((ext_vector_type(16))) __bf16 v16bf;
typedef __attribute__((ext_vector_type(8)))  float  v8f;

#define NN 512
#define DD 128
#define KMEM 384   // 2*D + DE

// ---------- helpers ----------
__device__ __forceinline__ unsigned short f32_to_bf16(float f) {
  unsigned int u = __float_as_uint(f);
  u += 0x7FFFu + ((u >> 16) & 1u);          // round-to-nearest-even
  return (unsigned short)(u >> 16);
}
__device__ __forceinline__ float bf16_to_f32(unsigned short h) {
  return __uint_as_float(((unsigned int)h) << 16);
}
__device__ __forceinline__ float wred_sum(float v) {
  #pragma unroll
  for (int m = 16; m >= 1; m >>= 1) v += __shfl_xor(v, m, 32);
  return v;
}

// swizzle (M,K32) -> A-fragment slot, contiguous K order (matches B loads):
// lanes 0-15: M=lane, elements hold K=0..15
// lanes 16-31: same M, K=16..31.
__device__ __forceinline__ void a_swizzle(int k32, int& haf, int& e) {
  haf = (k32 >> 4) & 1;
  int base = k32 - 16 * haf;
  e = base;
}

// K=128 GEMM on a pre-swizzled LDS A tile; B from bf16 transposed weights WT[col][k].
__device__ __forceinline__ void wmma_k128(const unsigned short* __restrict__ WT,
                                          const unsigned short* memA,
                                          int lane, int cb0, int cb1,
                                          v8f& acc0, v8f& acc1) {
  const int haf = lane >> 4, l16 = lane & 15;
  #pragma unroll
  for (int kc = 0; kc < 4; ++kc) {
    v16bf a  = *(const v16bf*)(memA + (kc * 32 + lane) * 16);
    v16bf b0 = *(const v16bf*)(WT + (cb0 + l16) * 128 + kc * 32 + haf * 16);
    v16bf b1 = *(const v16bf*)(WT + (cb1 + l16) * 128 + kc * 32 + haf * 16);
    acc0 = __builtin_amdgcn_wmma_f32_16x16x32_bf16(false, a, false, b0, (short)0, acc0, false, false);
    acc1 = __builtin_amdgcn_wmma_f32_16x16x32_bf16(false, a, false, b1, (short)0, acc1, false, false);
  }
}

// ---------- kernel 0: weight transpose + bf16 convert ----------
__global__ __launch_bounds__(256) void prep_weights(
    const float* __restrict__ W_mem, const float* __restrict__ W_e,
    const float* __restrict__ Wk, const float* __restrict__ Wv,
    unsigned short* __restrict__ WT_mem, unsigned short* __restrict__ WT_e,
    unsigned short* __restrict__ WT_k, unsigned short* __restrict__ WT_v) {
  int idx = blockIdx.x * blockDim.x + threadIdx.x;
  if (idx < 128 * KMEM) {                     // WT_mem[col][k]
    int c = idx / KMEM, k = idx - c * KMEM;
    WT_mem[idx] = f32_to_bf16(W_mem[k * 128 + c]);
    return;
  }
  int r = idx - 128 * KMEM;
  if (r < 3 * 128 * 128) {
    int which = r / 16384, p = r - which * 16384;
    int c = p / 128, k = p - c * 128;
    const float* W = (which == 0) ? W_e : (which == 1) ? Wk : Wv;
    unsigned short* WT = (which == 0) ? WT_e : (which == 1) ? WT_k : WT_v;
    WT[p] = f32_to_bf16(W[k * 128 + c]);
  }
}

// ---------- kernel 1: q = node @ Wq + bq ----------
__global__ __launch_bounds__(128) void q_proj(
    const float* __restrict__ node, const float* __restrict__ Wq,
    const float* __restrict__ bq, float* __restrict__ q_ws) {
  __shared__ float nrow[128];
  const int n = blockIdx.x, c = threadIdx.x;
  nrow[c] = node[n * 128 + c];
  __syncthreads();
  float acc = bq[c];
  #pragma unroll 4
  for (int k = 0; k < 128; ++k) acc += nrow[k] * Wq[k * 128 + c];
  q_ws[n * 128 + c] = acc;
}

// ---------- kernel 2: fused per-edge pipeline ----------
// block = (i, j0..j0+15): memory = relu(LN(concat@W_mem)); edge_out; scores; v.
__global__ __launch_bounds__(128) void fused_edge_kernel(
    const float* __restrict__ node, const float* __restrict__ edge,
    const unsigned char* __restrict__ edge_mask,
    const float* __restrict__ b_mem, const float* __restrict__ g_mem, const float* __restrict__ be_mem,
    const float* __restrict__ b_e,  const float* __restrict__ g_e1, const float* __restrict__ be_e1,
    const float* __restrict__ g_e2, const float* __restrict__ be_e2,
    const float* __restrict__ bk,   const float* __restrict__ bv,
    const unsigned short* __restrict__ WT_mem, const unsigned short* __restrict__ WT_e,
    const unsigned short* __restrict__ WT_k,   const unsigned short* __restrict__ WT_v,
    const float* __restrict__ q_ws, float* __restrict__ scores_ws,
    unsigned short* __restrict__ v_ws, float* __restrict__ out_edge) {
  __shared__ alignas(32) unsigned short A_lds[12 * 32 * 16];  // 12KB, A-frag swizzled
  __shared__ alignas(32) unsigned short memA[4 * 32 * 16];    // 4KB, memory tile bf16 swizzled
  __shared__ float edge_f32[16 * 128];                        // 8KB, residual copy
  __shared__ float tmp[16 * 128];                             // 8KB, GEMM dump buffer

  const int t = threadIdx.x;
  const int lane = t & 31, wave = t >> 5;
  const int haf = lane >> 4, l16 = lane & 15;
  const int i = blockIdx.x >> 5;            // key index m
  const int j0 = (blockIdx.x & 31) << 4;    // batch tile base n
  const int cb0 = wave * 32, cb1 = wave * 32 + 16;

  // ---- stage A tile [16 x 384] -> bf16, pre-swizzled; keep edge slice in f32 ----
  for (int idx = t; idx < 16 * KMEM; idx += 128) {
    int M = idx / KMEM, K = idx - M * KMEM;
    float val;
    if (K < 128) {
      val = edge[((size_t)i * NN + (j0 + M)) * 128 + K];
      edge_f32[M * 128 + K] = val;
    } else if (K < 256) {
      val = node[(j0 + M) * 128 + (K - 128)];   // src = node[j]
    } else {
      val = node[i * 128 + (K - 256)];          // tar = node[i]
    }
    int kc = K >> 5, hf, e;
    a_swizzle(K & 31, hf, e);
    A_lds[(kc * 32 + (M + 16 * hf)) * 16 + e] = f32_to_bf16(val);
  }
  __syncthreads();

  // ---- GEMM1: [16x384] @ W_mem -> tmp (+bias) ----
  {
    v8f acc0 = {}, acc1 = {};
    #pragma unroll
    for (int kc = 0; kc < 12; ++kc) {
      v16bf a  = *(const v16bf*)(A_lds + (kc * 32 + lane) * 16);
      v16bf b0 = *(const v16bf*)(WT_mem + (cb0 + l16) * KMEM + kc * 32 + haf * 16);
      v16bf b1 = *(const v16bf*)(WT_mem + (cb1 + l16) * KMEM + kc * 32 + haf * 16);
      acc0 = __builtin_amdgcn_wmma_f32_16x16x32_bf16(false, a, false, b0, (short)0, acc0, false, false);
      acc1 = __builtin_amdgcn_wmma_f32_16x16x32_bf16(false, a, false, b1, (short)0, acc1, false, false);
    }
    #pragma unroll
    for (int v = 0; v < 8; ++v) {           // C layout: lane 0-15 N=lane M=v; lane16-31 N=lane-16 M=v+8
      int M = v + 8 * haf;
      tmp[M * 128 + cb0 + l16] = acc0[v] + b_mem[cb0 + l16];
      tmp[M * 128 + cb1 + l16] = acc1[v] + b_mem[cb1 + l16];
    }
  }
  __syncthreads();

  // ---- LN + ReLU -> memory tile (bf16, swizzled) ----
  for (int rr = 0; rr < 4; ++rr) {
    int r = wave + 4 * rr;
    float x[4]; float s = 0.f, s2 = 0.f;
    #pragma unroll
    for (int p = 0; p < 4; ++p) {
      x[p] = tmp[r * 128 + lane + 32 * p];
      s += x[p]; s2 += x[p] * x[p];
    }
    s = wred_sum(s); s2 = wred_sum(s2);
    float mean = s * (1.f / 128.f);
    float rstd = rsqrtf(s2 * (1.f / 128.f) - mean * mean + 1e-5f);
    #pragma unroll
    for (int p = 0; p < 4; ++p) {
      int c = lane + 32 * p;
      float y = fmaxf((x[p] - mean) * rstd * g_mem[c] + be_mem[c], 0.f);
      int hf, e; a_swizzle(c & 31, hf, e);
      memA[((c >> 5) * 32 + (r + 16 * hf)) * 16 + e] = f32_to_bf16(y);
    }
  }
  __syncthreads();

  // ---- edge branch: LN(edge + relu(LN(memory@W_e + b_e))) ----
  {
    v8f e0 = {}, e1 = {};
    wmma_k128(WT_e, memA, lane, cb0, cb1, e0, e1);
    #pragma unroll
    for (int v = 0; v < 8; ++v) {
      int M = v + 8 * haf;
      tmp[M * 128 + cb0 + l16] = e0[v] + b_e[cb0 + l16];
      tmp[M * 128 + cb1 + l16] = e1[v] + b_e[cb1 + l16];
    }
  }
  __syncthreads();
  for (int rr = 0; rr < 4; ++rr) {
    int r = wave + 4 * rr;
    float x[4]; float s = 0.f, s2 = 0.f;
    #pragma unroll
    for (int p = 0; p < 4; ++p) {
      x[p] = tmp[r * 128 + lane + 32 * p];
      s += x[p]; s2 += x[p] * x[p];
    }
    s = wred_sum(s); s2 = wred_sum(s2);
    float mean = s * (1.f / 128.f);
    float rstd = rsqrtf(s2 * (1.f / 128.f) - mean * mean + 1e-5f);
    float y[4]; float u = 0.f, u2 = 0.f;
    #pragma unroll
    for (int p = 0; p < 4; ++p) {
      int c = lane + 32 * p;
      float z = fmaxf((x[p] - mean) * rstd * g_e1[c] + be_e1[c], 0.f);
      y[p] = z + edge_f32[r * 128 + c];
      u += y[p]; u2 += y[p] * y[p];
    }
    u = wred_sum(u); u2 = wred_sum(u2);
    float mean2 = u * (1.f / 128.f);
    float rstd2 = rsqrtf(u2 * (1.f / 128.f) - mean2 * mean2 + 1e-5f);
    #pragma unroll
    for (int p = 0; p < 4; ++p) {
      int c = lane + 32 * p;
      out_edge[((size_t)i * NN + (j0 + r)) * 128 + c] =
          (y[p] - mean2) * rstd2 * g_e2[c] + be_e2[c];
    }
  }
  __syncthreads();

  // ---- k branch: memory@Wk + bk, then per-head q.k scores ----
  {
    v8f k0 = {}, k1 = {};
    wmma_k128(WT_k, memA, lane, cb0, cb1, k0, k1);
    #pragma unroll
    for (int v = 0; v < 8; ++v) {
      int M = v + 8 * haf;
      tmp[M * 128 + cb0 + l16] = k0[v] + bk[cb0 + l16];
      tmp[M * 128 + cb1 + l16] = k1[v] + bk[cb1 + l16];
    }
  }
  __syncthreads();
  {
    int row = t >> 3, head = t & 7;          // 16 rows x 8 heads
    const float* qp = q_ws + (size_t)(j0 + row) * 128 + head * 16;
    const float* kp = tmp + row * 128 + head * 16;
    float s = 0.f;
    #pragma unroll
    for (int d = 0; d < 16; ++d) s += kp[d] * qp[d];
    s *= 0.25f;                               // 1/sqrt(16)
    if (edge_mask[(size_t)(j0 + row) * NN + i]) s = -__builtin_huge_valf();
    scores_ws[((size_t)(j0 + row) * 8 + head) * NN + i] = s;  // [n][h][m]
  }
  __syncthreads();

  // ---- v branch: memory@Wv + bv -> bf16 workspace [n][m][d] ----
  {
    v8f v0 = {}, v1 = {};
    wmma_k128(WT_v, memA, lane, cb0, cb1, v0, v1);
    #pragma unroll
    for (int v = 0; v < 8; ++v) {
      int M = v + 8 * haf;
      tmp[M * 128 + cb0 + l16] = v0[v] + bv[cb0 + l16];
      tmp[M * 128 + cb1 + l16] = v1[v] + bv[cb1 + l16];
    }
  }
  __syncthreads();
  for (int r = 0; r < 16; ++r)
    v_ws[((size_t)(j0 + r) * NN + i) * 128 + t] = f32_to_bf16(tmp[r * 128 + t]);
}

// ---------- kernel 3: softmax + attn@v + node pipeline ----------
__global__ __launch_bounds__(128) void attn_node_kernel(
    const float* __restrict__ node, const float* __restrict__ scores_ws,
    const unsigned short* __restrict__ v_ws,
    const float* __restrict__ Wo, const float* __restrict__ bo,
    const float* __restrict__ W1, const float* __restrict__ b1,
    const float* __restrict__ W2, const float* __restrict__ b2,
    const float* __restrict__ g2, const float* __restrict__ be2,
    const float* __restrict__ g3, const float* __restrict__ be3,
    float* __restrict__ out_x) {
  __shared__ float s_lds[8 * 512];   // 16KB scores -> attn
  __shared__ float obuf[128];
  __shared__ float x1buf[128];
  __shared__ float hbuf[256];
  __shared__ float red[8];

  const int n = blockIdx.x, t = threadIdx.x;
  const int lane = t & 31, wave = t >> 5;

  for (int idx = t; idx < 4096; idx += 128)
    s_lds[idx] = scores_ws[(size_t)n * 4096 + idx];
  __syncthreads();

  // softmax per head (16 threads per head, shfl width 16)
  {
    int head = t >> 4, sub = t & 15;
    float mx = -3.4e38f;
    for (int m = sub; m < 512; m += 16) mx = fmaxf(mx, s_lds[head * 512 + m]);
    #pragma unroll
    for (int msk = 8; msk >= 1; msk >>= 1) mx = fmaxf(mx, __shfl_xor(mx, msk, 16));
    float den = 0.f;
    for (int m = sub; m < 512; m += 16) den += __expf(s_lds[head * 512 + m] - mx);
    #pragma unroll
    for (int msk = 8; msk >= 1; msk >>= 1) den += __shfl_xor(den, msk, 16);
    float inv = 1.f / den;
    for (int m = sub; m < 512; m += 16)
      s_lds[head * 512 + m] = __expf(s_lds[head * 512 + m] - mx) * inv;
  }
  __syncthreads();

  // o[n, t] = sum_m attn[n, h(t), m] * v[m, n, t]
  {
    int head = t >> 4;
    const unsigned short* vp = v_ws + (size_t)n * 512 * 128 + t;
    const float* ap = s_lds + head * 512;
    float acc = 0.f;
    #pragma unroll 4
    for (int m = 0; m < 512; ++m) acc += ap[m] * bf16_to_f32(vp[(size_t)m * 128]);
    obuf[t] = acc;
  }
  __syncthreads();

  // x1 = LN(node + o @ Wo + bo)
  float oc = bo[t];
  #pragma unroll 4
  for (int k = 0; k < 128; ++k) oc += obuf[k] * Wo[k * 128 + t];
  float xr = node[n * 128 + t] + oc;
  {
    float s = wred_sum(xr), s2 = wred_sum(xr * xr);
    if (lane == 0) { red[wave] = s; red[4 + wave] = s2; }
    __syncthreads();
    float ts = red[0] + red[1] + red[2] + red[3];
    float ts2 = red[4] + red[5] + red[6] + red[7];
    float mean = ts * (1.f / 128.f);
    float rstd = rsqrtf(ts2 * (1.f / 128.f) - mean * mean + 1e-5f);
    __syncthreads();
    x1buf[t] = (xr - mean) * rstd * g2[t] + be2[t];
  }
  __syncthreads();

  // FFN: h = relu(x1 @ W1 + b1); f = h @ W2 + b2
  {
    float h0 = b1[t], h1v = b1[t + 128];
    #pragma unroll 4
    for (int k = 0; k < 128; ++k) {
      float xv = x1buf[k];
      h0  += xv * W1[k * 256 + t];
      h1v += xv * W1[k * 256 + t + 128];
    }
    hbuf[t] = fmaxf(h0, 0.f);
    hbuf[t + 128] = fmaxf(h1v, 0.f);
  }
  __syncthreads();
  float f = b2[t];
  #pragma unroll 4
  for (int k = 0; k < 256; ++k) f += hbuf[k] * W2[k * 128 + t];
  float y = x1buf[t] + f;
  {
    float s = wred_sum(y), s2 = wred_sum(y * y);
    if (lane == 0) { red[wave] = s; red[4 + wave] = s2; }
    __syncthreads();
    float ts = red[0] + red[1] + red[2] + red[3];
    float ts2 = red[4] + red[5] + red[6] + red[7];
    float mean = ts * (1.f / 128.f);
    float rstd = rsqrtf(ts2 * (1.f / 128.f) - mean * mean + 1e-5f);
    out_x[n * 128 + t] = (y - mean) * rstd * g3[t] + be3[t];
  }
}

// ---------- launch ----------
extern "C" void kernel_launch(void* const* d_in, const int* in_sizes, int n_in,
                              void* d_out, int out_size, void* d_ws, size_t ws_size,
                              hipStream_t stream) {
  const float* node      = (const float*)d_in[0];
  const float* edge      = (const float*)d_in[1];
  const unsigned char* edge_mask = (const unsigned char*)d_in[2];
  const float* W_mem = (const float*)d_in[3];
  const float* b_mem = (const float*)d_in[4];
  const float* g_mem = (const float*)d_in[5];
  const float* be_mem = (const float*)d_in[6];
  const float* W_e  = (const float*)d_in[7];
  const float* b_e  = (const float*)d_in[8];
  const float* g_e1 = (const float*)d_in[9];
  const float* be_e1 = (const float*)d_in[10];
  const float* g_e2 = (const float*)d_in[11];
  const float* be_e2 = (const float*)d_in[12];
  const float* Wq = (const float*)d_in[13];
  const float* bq = (const float*)d_in[14];
  const float* Wk = (const float*)d_in[15];
  const float* bk = (const float*)d_in[16];
  const float* Wv = (const float*)d_in[17];
  const float* bv = (const float*)d_in[18];
  const float* Wo = (const float*)d_in[19];
  const float* bo = (const float*)d_in[20];
  const float* W1 = (const float*)d_in[21];
  const float* b1 = (const float*)d_in[22];
  const float* W2 = (const float*)d_in[23];
  const float* b2 = (const float*)d_in[24];
  const float* g2 = (const float*)d_in[25];
  const float* be2 = (const float*)d_in[26];
  const float* g3 = (const float*)d_in[27];
  const float* be3 = (const float*)d_in[28];

  float* out_x = (float*)d_out;
  float* out_edge = out_x + 512 * 128;

  // workspace layout (byte offsets, all 256-aligned)
  char* ws = (char*)d_ws;
  const size_t OFF_Q      = 0;                       // 512*128 f32      = 256 KB
  const size_t OFF_SCORES = OFF_Q + 262144;          // 512*8*512 f32    = 8 MB
  const size_t OFF_WTMEM  = OFF_SCORES + 8388608;    // 128*384 bf16     = 96 KB
  const size_t OFF_WTE    = OFF_WTMEM + 98304;       // 128*128 bf16     = 32 KB
  const size_t OFF_WTK    = OFF_WTE + 32768;
  const size_t OFF_WTV    = OFF_WTK + 32768;
  const size_t OFF_V      = OFF_WTV + 32768;         // 512*512*128 bf16 = 64 MB

  float* q_ws       = (float*)(ws + OFF_Q);
  float* scores_ws  = (float*)(ws + OFF_SCORES);
  unsigned short* WT_mem = (unsigned short*)(ws + OFF_WTMEM);
  unsigned short* WT_e   = (unsigned short*)(ws + OFF_WTE);
  unsigned short* WT_k   = (unsigned short*)(ws + OFF_WTK);
  unsigned short* WT_v   = (unsigned short*)(ws + OFF_WTV);
  unsigned short* v_ws   = (unsigned short*)(ws + OFF_V);

  prep_weights<<<384, 256, 0, stream>>>(W_mem, W_e, Wk, Wv, WT_mem, WT_e, WT_k, WT_v);
  q_proj<<<512, 128, 0, stream>>>(node, Wq, bq, q_ws);
  fused_edge_kernel<<<512 * 32, 128, 0, stream>>>(
      node, edge, edge_mask,
      b_mem, g_mem, be_mem, b_e, g_e1, be_e1, g_e2, be_e2, bk, bv,
      WT_mem, WT_e, WT_k, WT_v, q_ws, scores_ws, v_ws, out_edge);
  attn_node_kernel<<<512, 128, 0, stream>>>(
      node, scores_ws, v_ws, Wo, bo, W1, b1, W2, b2, g2, be2, g3, be3, out_x);
}
